// LengthRegulator_83657372992041
// MI455X (gfx1250) — matrix operand order repaired and checked by reading, not compiled, not run on hardware
//
#include <hip/hip_runtime.h>
#include <stdint.h>
#include <stddef.h>

// Problem constants (from reference)
#define NN   16
#define LL   512
#define DD   256
#define FF   256
#define MEL  3584          // L * (MAXDUR-1)
#define LP   (LL + 2)      // halo-padded length (rows -1 and L zeroed)
#define KSTEPS 24          // 3*256 / 32
#define CTILES 16          // 256 / 16
#define LN_EPS 1e-5f

typedef __attribute__((ext_vector_type(16))) _Float16 v16h;
typedef __attribute__((ext_vector_type(8)))  float    v8f;

// ---------------------------------------------------------------------------
// Prep: cast x (f32) -> padded f16 [NN][LP][DD]; also zero h1's halo rows.
// ---------------------------------------------------------------------------
__global__ void prep_x(const float* __restrict__ x,
                       _Float16* __restrict__ xh,
                       _Float16* __restrict__ h1h) {
    int idx = blockIdx.x * 256 + threadIdx.x;
    const int tot = NN * LP * DD;
    if (idx >= tot) return;
    int ci = idx % DD;
    int lp = (idx / DD) % LP;
    int n  = idx / (DD * LP);
    _Float16 v = (_Float16)0.0f;
    if (lp >= 1 && lp <= LL) {
        v = (_Float16)x[((size_t)n * LL + (lp - 1)) * DD + ci];
    } else {
        h1h[idx] = (_Float16)0.0f;   // zero halo of the hidden buffer too
    }
    xh[idx] = v;
}

// ---------------------------------------------------------------------------
// Prep: repack conv weights [3][256][256] f32 into WMMA B-fragment order f16.
// Layout: frag[((ks*16 + ct)*32 + lane)*16 + i]
//   lane%16 = column-within-tile, lane/16 = khi (K half: 0..15 vs 16..31)
//   i -> VGPR pair p=i/2, hi=i%2 ; kloc = khi*16 + 2p + hi   (contiguous K, same as A)
// ---------------------------------------------------------------------------
__global__ void prep_wfrag(const float* __restrict__ w,
                           _Float16* __restrict__ frag) {
    int idx = blockIdx.x * 256 + threadIdx.x;
    if (idx >= KSTEPS * CTILES * 32 * 16) return;
    int i    = idx & 15;
    int lane = (idx >> 4) & 31;
    int ct   = (idx >> 9) & 15;
    int ks   = idx >> 13;
    int nc = lane & 15, khi = lane >> 4;
    int p = i >> 1, hi = i & 1;
    int kloc = khi * 16 + 2 * p + hi;
    int kg   = ks * 32 + kloc;          // 0..767
    int ci   = kg & 255;
    int tap  = kg >> 8;                 // 0..2
    int co   = ct * 16 + nc;
    frag[idx] = (_Float16)w[((size_t)tap * DD + ci) * FF + co];
}

// ---------------------------------------------------------------------------
// Fused conv1d(K=3) as 768-deep WMMA GEMM + bias + ReLU + LayerNorm.
// mode 0: write f16 hidden (padded) for the next conv.
// mode 1: fuse the 256->1 linear and write log_dur.
// Grid: NN * (LL/32) = 256 blocks, 256 threads (8 waves).
// Wave w: subrow = w/4 (which 16-row half), cg = w%4 (which 64-col group).
// ---------------------------------------------------------------------------
__global__ __launch_bounds__(256)
void vp_conv_ln(const _Float16* __restrict__ xin,     // [NN][LP][DD] f16
                const _Float16* __restrict__ wfrag,   // B fragments
                const float*    __restrict__ bias,
                const float*    __restrict__ lng,
                const float*    __restrict__ lnb,
                _Float16*       __restrict__ hout,    // mode 0
                const float*    __restrict__ linw,    // mode 1
                const float*    __restrict__ linb,    // mode 1
                float*          __restrict__ logdur,  // mode 1
                int mode) {
    __shared__ __align__(16) _Float16 Xs[34 * DD];    // 32 rows + 2 halo
    __shared__ float Cs[32 * DD];
    __shared__ float Red[32][8];
    __shared__ float Red2[32][8];
    __shared__ float Stat[32][2];

    const int tid    = threadIdx.x;
    const int n      = blockIdx.x >> 4;
    const int lt     = blockIdx.x & 15;
    const int lane   = tid & 31;
    const int wv     = tid >> 5;
    const int subrow = wv >> 2;
    const int cg     = wv & 3;

    // Stage 34 padded input rows (lp = lt*32 .. lt*32+33) into LDS, 16B chunks
    {
        const uint4* src = (const uint4*)(xin + ((size_t)n * LP + (size_t)lt * 32) * DD);
        uint4* dst = (uint4*)Xs;
        for (int i = tid; i < (34 * DD) / 8; i += 256) dst[i] = src[i];
    }
    __syncthreads();

    v8f acc[4];
#pragma unroll
    for (int t = 0; t < 4; ++t)
#pragma unroll
        for (int v = 0; v < 8; ++v) acc[t][v] = 0.0f;

    const int m   = lane & 15;
    const int khi = lane >> 4;

    for (int ks = 0; ks < KSTEPS; ++ks) {
        const int tap = ks >> 3;               // which conv tap this k-chunk is
        const int cib = (ks & 7) * 32;         // ci base within the tap
        const int r   = subrow * 16 + m + tap; // LDS row (0..33)
        const uint32_t* xr = (const uint32_t*)(Xs + r * DD + cib);

        union { v16h h; uint32_t u[8]; } A;
#pragma unroll
        for (int p = 0; p < 8; ++p) {
            // A layout matching B fragments: VGPR p -> K khi*16 + 2p, +1
            int k = khi * 16 + (p << 1);
            A.u[p] = xr[k >> 1];
        }

#pragma unroll
        for (int t = 0; t < 4; ++t) {
            const int ct = cg * 4 + t;
            const v16h B = *(const v16h*)(wfrag + (((size_t)ks * 16 + ct) * 32 + lane) * 16);
            acc[t] = __builtin_amdgcn_wmma_f32_16x16x32_f16(
                false, A.h, false, B, (short)0, acc[t], false, false);
        }
    }

    // bias + ReLU into LDS (C layout: col = lane%16, rows v (+8 for hi lanes))
    {
        const int c     = lane & 15;
        const int rbase = subrow * 16 + ((lane >> 4) << 3);
#pragma unroll
        for (int t = 0; t < 4; ++t) {
            const int co = cg * 64 + t * 16 + c;
            const float bb = bias[co];
#pragma unroll
            for (int v = 0; v < 8; ++v)
                Cs[(rbase + v) * DD + co] = fmaxf(acc[t][v] + bb, 0.0f);
        }
    }
    __syncthreads();

    // LayerNorm stats: 8 partial segments per row
    {
        const int row = tid >> 3, seg = tid & 7;
        const float* cr = Cs + row * DD + seg * 32;
        float s = 0.0f, sq = 0.0f;
#pragma unroll
        for (int c = 0; c < 32; ++c) { float v = cr[c]; s += v; sq += v * v; }
        Red[row][seg] = s; Red2[row][seg] = sq;
    }
    __syncthreads();
    if (tid < 32) {
        float s = 0.0f, sq = 0.0f;
#pragma unroll
        for (int g = 0; g < 8; ++g) { s += Red[tid][g]; sq += Red2[tid][g]; }
        const float mu  = s * (1.0f / DD);
        const float var = sq * (1.0f / DD) - mu * mu;
        Stat[tid][0] = mu;
        Stat[tid][1] = rsqrtf(var + LN_EPS);
    }
    __syncthreads();

    if (mode == 0) {
        // normalized f16 hidden, coalesced column-per-thread stores
        const float gc = lng[tid];
        const float bc = lnb[tid];
        for (int row = 0; row < 32; ++row) {
            const float mu = Stat[row][0], rs = Stat[row][1];
            const float v = (Cs[row * DD + tid] - mu) * rs * gc + bc;
            hout[(((size_t)n * LP) + (size_t)lt * 32 + row + 1) * DD + tid] = (_Float16)v;
        }
    } else {
        // fuse LN * lin_w dot product -> log_dur
        const int row = tid >> 3, seg = tid & 7;
        const float mu = Stat[row][0], rs = Stat[row][1];
        const float* cr = Cs + row * DD;
        float dp = 0.0f;
        for (int c = seg * 32; c < seg * 32 + 32; ++c) {
            const float v = (cr[c] - mu) * rs * lng[c] + lnb[c];
            dp += v * linw[c];
        }
        Red[row][seg] = dp;
        __syncthreads();
        if (tid < 32) {
            float s = 0.0f;
#pragma unroll
            for (int g = 0; g < 8; ++g) s += Red[tid][g];
            logdur[(size_t)n * LL + lt * 32 + tid] = s + linb[0];
        }
    }
}

// ---------------------------------------------------------------------------
// Length regulator part 1: cumsum durations + binary-search row map.
// ---------------------------------------------------------------------------
__global__ void dur_scan_map(const int* __restrict__ target,
                             int* __restrict__ rowmap) {
    __shared__ int ends[LL];
    __shared__ int total;
    const int n = blockIdx.x, tid = threadIdx.x;
    for (int i = tid; i < LL; i += 256) ends[i] = target[n * LL + i];
    __syncthreads();
    if (tid == 0) {
        int s = 0;
        for (int i = 0; i < LL; ++i) { s += ends[i]; ends[i] = s; }
        total = s;
    }
    __syncthreads();
    const int tot = total;
    for (int mm = tid; mm < MEL; mm += 256) {
        int res = -1;
        if (mm < tot) {
            int lo = 0, hi = LL - 1;                // first l with ends[l] > mm
            while (lo < hi) { int mid = (lo + hi) >> 1; if (ends[mid] > mm) hi = mid; else lo = mid + 1; }
            res = lo;
        }
        rowmap[(size_t)n * MEL + mm] = res;
    }
}

// ---------------------------------------------------------------------------
// Length regulator part 2: gather x rows (or zeros) into [NN][MEL][DD].
// One 64-thread block per output row; float4 per thread -> fully coalesced.
// ---------------------------------------------------------------------------
__global__ void gather_rows(const float* __restrict__ x,
                            const int* __restrict__ rowmap,
                            float* __restrict__ out) {
    const int row = blockIdx.x;                  // 0..NN*MEL-1
    const int n = row / MEL;
    const int l = rowmap[row];
    float4* o = (float4*)(out + (size_t)row * DD);
    const int t = threadIdx.x;                   // 0..63
    if (l >= 0) {
        const float4* s = (const float4*)(x + ((size_t)n * LL + l) * DD);
        o[t] = s[t];
    } else {
        o[t] = make_float4(0.0f, 0.0f, 0.0f, 0.0f);
    }
}

// ---------------------------------------------------------------------------
extern "C" void kernel_launch(void* const* d_in, const int* in_sizes, int n_in,
                              void* d_out, int out_size, void* d_ws, size_t ws_size,
                              hipStream_t stream) {
    const float* x       = (const float*)d_in[0];
    const float* conv1_w = (const float*)d_in[1];
    const float* conv1_b = (const float*)d_in[2];
    const float* ln1_g   = (const float*)d_in[3];
    const float* ln1_b   = (const float*)d_in[4];
    const float* conv2_w = (const float*)d_in[5];
    const float* conv2_b = (const float*)d_in[6];
    const float* ln2_g   = (const float*)d_in[7];
    const float* ln2_b   = (const float*)d_in[8];
    const float* lin_w   = (const float*)d_in[9];
    const float* lin_b   = (const float*)d_in[10];
    const int*   target  = (const int*)d_in[11];

    float* out    = (float*)d_out;                       // [NN][MEL][DD]
    float* logdur = out + (size_t)NN * MEL * DD;         // [NN][LL]

    // Workspace carve-up (16B-aligned offsets), total ~9 MB
    char* ws = (char*)d_ws;
    const size_t xh_bytes = (size_t)NN * LP * DD * sizeof(_Float16);   // 4,210,688
    const size_t wf_bytes = (size_t)KSTEPS * CTILES * 32 * 16 * sizeof(_Float16); // 393,216
    _Float16* xh  = (_Float16*)(ws);
    _Float16* h1  = (_Float16*)(ws + xh_bytes);
    _Float16* w1f = (_Float16*)(ws + 2 * xh_bytes);
    _Float16* w2f = (_Float16*)(ws + 2 * xh_bytes + wf_bytes);
    int*      map = (int*)     (ws + 2 * xh_bytes + 2 * wf_bytes);

    const int totx = NN * LP * DD;
    prep_x<<<(totx + 255) / 256, 256, 0, stream>>>(x, xh, h1);

    const int totw = KSTEPS * CTILES * 32 * 16;
    prep_wfrag<<<(totw + 255) / 256, 256, 0, stream>>>(conv1_w, w1f);
    prep_wfrag<<<(totw + 255) / 256, 256, 0, stream>>>(conv2_w, w2f);

    // conv1 + ReLU + LN1 -> h1 (f16, padded)
    vp_conv_ln<<<NN * (LL / 32), 256, 0, stream>>>(
        xh, w1f, conv1_b, ln1_g, ln1_b, h1, nullptr, nullptr, nullptr, 0);

    // conv2 + ReLU + LN2 + linear -> log_dur
    vp_conv_ln<<<NN * (LL / 32), 256, 0, stream>>>(
        h1, w2f, conv2_b, ln2_g, ln2_b, nullptr, lin_w, lin_b, logdur, 1);

    // length regulator
    dur_scan_map<<<NN, 256, 0, stream>>>(target, map);
    gather_rows<<<NN * MEL, 64, 0, stream>>>(x, map, out);
}
